// MultiHeadCrossAttentionV2_50465865728154
// MI455X (gfx1250) — matrix operand order repaired and checked by reading, not compiled, not run on hardware
//
#include <hip/hip_runtime.h>

#define DEV __device__ __forceinline__

typedef __attribute__((ext_vector_type(16))) __bf16 bf16x16;
typedef __attribute__((ext_vector_type(8)))  float  f32x8;
typedef unsigned int u32x4 __attribute__((ext_vector_type(4)));
typedef int          i32x8 __attribute__((ext_vector_type(8)));
typedef int          i32x4 __attribute__((ext_vector_type(4)));

union FragU { bf16x16 v; uint4 q[2]; };

DEV unsigned short f2bf(float f) {
  unsigned int u = __float_as_uint(f);
  unsigned int r = u + 0x7FFFu + ((u >> 16) & 1u);
  return (unsigned short)(r >> 16);
}

DEV bf16x16 load_frag(const unsigned short* p0, const unsigned short* p1) {
  FragU u;
  u.q[0] = *reinterpret_cast<const uint4*>(p0);
  u.q[1] = *reinterpret_cast<const uint4*>(p1);
  return u.v;
}

DEV f32x8 wmma_bf16(bf16x16 a, bf16x16 b, f32x8 c) {
  return __builtin_amdgcn_wmma_f32_16x16x32_bf16(false, a, false, b, (short)0, c,
                                                 false, false);
}

DEV f32x8 zero8() {
  f32x8 z;
#pragma unroll
  for (int i = 0; i < 8; ++i) z[i] = 0.0f;
  return z;
}

// Generic->LDS address: flat aperture maps LDS via addr[31:0] (ISA 10.2), so
// truncation of the generic pointer yields the LDS byte address.
DEV unsigned lds_addr32(const void* p) { return (unsigned)(uintptr_t)p; }

// XOR-shuffle within 16-lane halves (the WMMA C-tile column group).
#define SWZ_XOR(x, m) \
  __int_as_float(__builtin_amdgcn_ds_swizzle(__float_as_int(x), ((m) << 10) | 0x1f))

// ---------------------------------------------------------------------------
// fp32 -> bf16 (RNE), 4 elements / thread
// ---------------------------------------------------------------------------
__global__ __launch_bounds__(256)
void cvt_f32_bf16_kernel(const float* __restrict__ src,
                         unsigned short* __restrict__ dst, int n4) {
  int i = blockIdx.x * 256 + threadIdx.x;
  if (i >= n4) return;
  float4 f = reinterpret_cast<const float4*>(src)[i];
  uint2 o;
  o.x = (unsigned)f2bf(f.x) | ((unsigned)f2bf(f.y) << 16);
  o.y = (unsigned)f2bf(f.z) | ((unsigned)f2bf(f.w) << 16);
  reinterpret_cast<uint2*>(dst)[i] = o;
}

// ---------------------------------------------------------------------------
// Y = X[M,1024] @ W[1024,1024]^T + bias. Weight tile staged into LDS via
// GLOBAL_LOAD_ASYNC_TO_LDS (ASYNCcnt path), consumed with ds_load_b128.
// Block: 256 thr = 8 waves; tile 128(M) x 64(N); wave computes 16x64.
// MODE 0: bf16 head-split [B,H,rows,64]; MODE 1: bf16 V^T [B,H,64,rows];
// MODE 2: fp32 flat [M,1024]. rpbShift = log2(rows per batch).
// ---------------------------------------------------------------------------
template<int MODE>
__global__ __launch_bounds__(256)
void gemm_xwT_kernel(const unsigned short* __restrict__ X,
                     const unsigned short* __restrict__ W,
                     const float* __restrict__ bias,
                     void* __restrict__ Y, int rpbShift) {
  constexpr int K = 1024;
  __shared__ __align__(16) unsigned short sB[64 * 40];  // 64 n-rows, stride 40

  const int lane = threadIdx.x & 31;
  const int wave = threadIdx.x >> 5;
  const int rlo  = lane & 15;
  const int kh   = lane >> 4;
  const int n0 = blockIdx.x * 64;
  const int m0 = blockIdx.y * 128 + wave * 16;

  const unsigned short* arow = X + (size_t)(m0 + rlo) * K;
  const int ldRow = threadIdx.x >> 2;
  const int ldKq  = (threadIdx.x & 3) * 8;
  const unsigned short* wsrc = W + (size_t)(n0 + ldRow) * K + ldKq;
  const unsigned sBdst = lds_addr32(&sB[ldRow * 40 + ldKq]);

  f32x8 acc[4];
#pragma unroll
  for (int t = 0; t < 4; ++t) acc[t] = zero8();

  for (int k0 = 0; k0 < K; k0 += 32) {
    // async DMA of W[n0..n0+63][k0..k0+31] into LDS (16B per lane)
    asm volatile("global_load_async_to_lds_b128 %0, %1, off"
                 :: "v"(sBdst), "v"(wsrc + k0) : "memory");
    asm volatile("s_wait_asynccnt 0x0" ::: "memory");
    __syncthreads();

    bf16x16 af = load_frag(arow + k0 + kh * 8, arow + k0 + 16 + kh * 8);
#pragma unroll
    for (int t = 0; t < 4; ++t) {
      const unsigned short* bp = &sB[(t * 16 + rlo) * 40 + kh * 8];
      acc[t] = wmma_bf16(af, load_frag(bp, bp + 16), acc[t]);
    }
    __syncthreads();
  }

#pragma unroll
  for (int t = 0; t < 4; ++t) {
#pragma unroll
    for (int r = 0; r < 8; ++r) {
      const int m = m0 + r + kh * 8;
      const int n = n0 + t * 16 + rlo;
      const float v = acc[t][r] + bias[n];
      if (MODE == 2) {
        reinterpret_cast<float*>(Y)[(size_t)m * 1024 + n] = v;
      } else {
        const int bb  = m >> rpbShift;
        const int row = m & ((1 << rpbShift) - 1);
        const int hh  = n >> 6;
        const int dd  = n & 63;
        unsigned short* o = reinterpret_cast<unsigned short*>(Y);
        if (MODE == 0)
          o[(((size_t)(bb * 16 + hh) << rpbShift) + row) * 64 + dd] = f2bf(v);
        else
          o[((size_t)(bb * 16 + hh) * 64 + dd) * ((size_t)1 << rpbShift) + row] =
              f2bf(v);
      }
    }
  }
}

// ---------------------------------------------------------------------------
// Flash attention. K/V tiles DMA'd into LDS by the Tensor Data Mover with
// descriptor padding (32-DWORD interval, 4-DWORD pad) => 72-element row
// stride, bank-conflict-free ds_load_b128 fragment reads.
// Grid (Nq/128, H, B); 8 waves x 16 q-rows; online softmax; V^T layout.
// ---------------------------------------------------------------------------
__global__ __launch_bounds__(256)
void flash_attn_kernel(const unsigned short* __restrict__ Qb,
                       const unsigned short* __restrict__ Kb,
                       const unsigned short* __restrict__ Vt,
                       const float* __restrict__ temperature,
                       unsigned short* __restrict__ Ob) {
  __shared__ __align__(16) unsigned short Ktile[64 * 72];
  __shared__ __align__(16) unsigned short Vtile[64 * 72];
  __shared__ __align__(16) unsigned short Pbuf[8][16 * 72];

  const int lane = threadIdx.x & 31;
  const int wave = threadIdx.x >> 5;
  const int rlo  = lane & 15;
  const int kh   = lane >> 4;
  const int h = blockIdx.y, b = blockIdx.z;
  const int q0 = blockIdx.x * 128 + wave * 16;
  const float kf = (0.125f / temperature[h]) * 1.44269504088896341f;

  const unsigned short* Qp = Qb + ((size_t)(b * 16 + h) * 1024 + q0) * 64;
  const unsigned short* Kp = Kb + (size_t)(b * 16 + h) * 2048 * 64;
  const unsigned short* Vp = Vt + (size_t)(b * 16 + h) * 64 * 2048;
  unsigned short* Pw = &Pbuf[wave][0];

  bf16x16 qf[2];
#pragma unroll
  for (int s = 0; s < 2; ++s) {
    const unsigned short* qp = Qp + rlo * 64 + s * 32 + kh * 8;
    qf[s] = load_frag(qp, qp + 16);
  }

  f32x8 o[4];
#pragma unroll
  for (int t = 0; t < 4; ++t) o[t] = zero8();
  float mrow[8], lrow[8];
#pragma unroll
  for (int r = 0; r < 8; ++r) { mrow[r] = -3.0e38f; lrow[r] = 0.0f; }

  // D# group-1 flags: data_size=2B, pad_enable, pad_interval=32DW, pad_amount=4DW
  const int g1flags = (1 << 16) | (1 << 20) | (4 << 22) | (3 << 25);
  const unsigned ldsK = lds_addr32(&Ktile[0]);
  const unsigned ldsV = lds_addr32(&Vtile[0]);

  for (int kv0 = 0; kv0 < 2048; kv0 += 64) {
#if __has_builtin(__builtin_amdgcn_tensor_load_to_lds)
    if (wave == 0) {
      const i32x4 gz  = {0, 0, 0, 0};
      const i32x8 gz8 = {0, 0, 0, 0, 0, 0, 0, 0};
      // K tile: 64 rows x 64 elem, row stride 64, rows kv0..kv0+63
      {
        unsigned long long ga =
            (unsigned long long)(uintptr_t)Kp + (unsigned long long)kv0 * 128ull;
        u32x4 g0 = {1u, ldsK, (unsigned)ga,
                    (unsigned)((ga >> 32) & 0x1FFFFFFu) | (2u << 30)};
        i32x8 g1 = {g1flags, 64 << 16, (int)(2048u << 16), 64 << 16,
                    64,      64,       0,                  0};
        __builtin_amdgcn_tensor_load_to_lds(g0, g1, gz, gz, gz8, 0);
      }
      // V tile: 64 rows (d) x 64 elem (kv), row stride 2048, col offset kv0
      {
        unsigned long long ga =
            (unsigned long long)(uintptr_t)Vp + (unsigned long long)kv0 * 2ull;
        u32x4 g0 = {1u, ldsV, (unsigned)ga,
                    (unsigned)((ga >> 32) & 0x1FFFFFFu) | (2u << 30)};
        i32x8 g1 = {g1flags, (int)(2048u << 16), 64 << 16, 64 << 16,
                    64,      2048,               0,        0};
        __builtin_amdgcn_tensor_load_to_lds(g0, g1, gz, gz, gz8, 0);
      }
      __builtin_amdgcn_s_wait_tensorcnt(0);
    }
#else
    for (int idx = threadIdx.x; idx < 64 * 8; idx += 256) {
      const int row = idx >> 3, c = (idx & 7) * 8;
      *reinterpret_cast<uint4*>(&Ktile[row * 72 + c]) =
          *reinterpret_cast<const uint4*>(Kp + (size_t)(kv0 + row) * 64 + c);
      *reinterpret_cast<uint4*>(&Vtile[row * 72 + c]) =
          *reinterpret_cast<const uint4*>(Vp + (size_t)row * 2048 + kv0 + c);
    }
#endif
    __syncthreads();

    // ---- S = Q @ K^T (16x64 per wave), K fragments from LDS -------------
    f32x8 sacc[4];
#pragma unroll
    for (int t = 0; t < 4; ++t) sacc[t] = zero8();
#pragma unroll
    for (int t = 0; t < 4; ++t) {
#pragma unroll
      for (int s = 0; s < 2; ++s) {
        const unsigned short* kpp = &Ktile[(t * 16 + rlo) * 72 + s * 32 + kh * 8];
        sacc[t] = wmma_bf16(qf[s], load_frag(kpp, kpp + 16), sacc[t]);
      }
    }

    // ---- online softmax --------------------------------------------------
    float mnew[8];
#pragma unroll
    for (int r = 0; r < 8; ++r) {
      float v = mrow[r];
#pragma unroll
      for (int t = 0; t < 4; ++t) v = fmaxf(v, sacc[t][r]);
      v = fmaxf(v, SWZ_XOR(v, 1));
      v = fmaxf(v, SWZ_XOR(v, 2));
      v = fmaxf(v, SWZ_XOR(v, 4));
      v = fmaxf(v, SWZ_XOR(v, 8));
      mnew[r] = v;
    }

    float alpha[8], rs[8];
#pragma unroll
    for (int r = 0; r < 8; ++r) {
      alpha[r] = exp2f((mrow[r] - mnew[r]) * kf);
      mrow[r]  = mnew[r];
      rs[r]    = 0.0f;
    }

#pragma unroll
    for (int t = 0; t < 4; ++t) {
#pragma unroll
      for (int r = 0; r < 8; ++r) {
        float p = exp2f((sacc[t][r] - mnew[r]) * kf);
        rs[r] += p;
        Pw[(r + kh * 8) * 72 + t * 16 + rlo] = f2bf(p);
      }
    }

#pragma unroll
    for (int r = 0; r < 8; ++r) {
      float v = rs[r];
      v += SWZ_XOR(v, 1);
      v += SWZ_XOR(v, 2);
      v += SWZ_XOR(v, 4);
      v += SWZ_XOR(v, 8);
      lrow[r] = lrow[r] * alpha[r] + v;
    }

#pragma unroll
    for (int t = 0; t < 4; ++t)
#pragma unroll
      for (int r = 0; r < 8; ++r) o[t][r] *= alpha[r];

    // ---- O += P @ V, both operands from LDS ------------------------------
#pragma unroll
    for (int s = 0; s < 2; ++s) {
      const unsigned short* pp = Pw + rlo * 72 + s * 32 + kh * 8;
      bf16x16 pf = load_frag(pp, pp + 16);
#pragma unroll
      for (int t = 0; t < 4; ++t) {
        const unsigned short* vpp = &Vtile[(t * 16 + rlo) * 72 + s * 32 + kh * 8];
        o[t] = wmma_bf16(pf, load_frag(vpp, vpp + 16), o[t]);
      }
    }
    __syncthreads();  // before next tile overwrites K/V staging
  }

#pragma unroll
  for (int r = 0; r < 8; ++r) lrow[r] = 1.0f / lrow[r];
#pragma unroll
  for (int t = 0; t < 4; ++t) {
#pragma unroll
    for (int r = 0; r < 8; ++r) {
      const int qi = q0 + r + kh * 8;
      const int d  = t * 16 + rlo;
      Ob[((size_t)(b * 1024 + qi)) * 1024 + h * 64 + d] = f2bf(o[t][r] * lrow[r]);
    }
  }
}

// ---------------------------------------------------------------------------
extern "C" void kernel_launch(void* const* d_in, const int* in_sizes, int n_in,
                              void* d_out, int out_size, void* d_ws, size_t ws_size,
                              hipStream_t stream) {
  (void)in_sizes; (void)n_in; (void)out_size; (void)ws_size;
  const float* q    = (const float*)d_in[0];
  const float* k    = (const float*)d_in[1];
  const float* v    = (const float*)d_in[2];
  const float* Wq   = (const float*)d_in[3];
  const float* bq   = (const float*)d_in[4];
  const float* Wk   = (const float*)d_in[5];
  const float* bk   = (const float*)d_in[6];
  const float* Wv   = (const float*)d_in[7];
  const float* bv   = (const float*)d_in[8];
  const float* Wo   = (const float*)d_in[9];
  const float* bo   = (const float*)d_in[10];
  const float* temp = (const float*)d_in[11];

  unsigned short* w = (unsigned short*)d_ws;
  unsigned short* Xq  = w;  w += (size_t)2 * 1024 * 1024;
  unsigned short* Xk  = w;  w += (size_t)2 * 2048 * 1024;
  unsigned short* Xv  = w;  w += (size_t)2 * 2048 * 1024;
  unsigned short* Wqb = w;  w += (size_t)1024 * 1024;
  unsigned short* Wkb = w;  w += (size_t)1024 * 1024;
  unsigned short* Wvb = w;  w += (size_t)1024 * 1024;
  unsigned short* Wob = w;  w += (size_t)1024 * 1024;
  unsigned short* Qbuf = w; w += (size_t)2 * 16 * 1024 * 64;
  unsigned short* Kbuf = w; w += (size_t)2 * 16 * 2048 * 64;
  unsigned short* Vtb  = w; w += (size_t)2 * 16 * 2048 * 64;
  unsigned short* Obuf = w; w += (size_t)2 * 1024 * 1024;

  cvt_f32_bf16_kernel<<<2048, 256, 0, stream>>>(q,  Xq,  2 * 1024 * 1024 / 4);
  cvt_f32_bf16_kernel<<<4096, 256, 0, stream>>>(k,  Xk,  2 * 2048 * 1024 / 4);
  cvt_f32_bf16_kernel<<<4096, 256, 0, stream>>>(v,  Xv,  2 * 2048 * 1024 / 4);
  cvt_f32_bf16_kernel<<<1024, 256, 0, stream>>>(Wq, Wqb, 1024 * 1024 / 4);
  cvt_f32_bf16_kernel<<<1024, 256, 0, stream>>>(Wk, Wkb, 1024 * 1024 / 4);
  cvt_f32_bf16_kernel<<<1024, 256, 0, stream>>>(Wv, Wvb, 1024 * 1024 / 4);
  cvt_f32_bf16_kernel<<<1024, 256, 0, stream>>>(Wo, Wob, 1024 * 1024 / 4);

  gemm_xwT_kernel<0><<<dim3(16, 16), 256, 0, stream>>>(Xq, Wqb, bq, Qbuf, 10);
  gemm_xwT_kernel<0><<<dim3(16, 32), 256, 0, stream>>>(Xk, Wkb, bk, Kbuf, 11);
  gemm_xwT_kernel<1><<<dim3(16, 32), 256, 0, stream>>>(Xv, Wvb, bv, Vtb, 11);

  flash_attn_kernel<<<dim3(8, 16, 2), 256, 0, stream>>>(Qbuf, Kbuf, Vtb, temp,
                                                        Obuf);

  gemm_xwT_kernel<2><<<dim3(16, 16), 256, 0, stream>>>(Obuf, Wob, bo, d_out, 10);
}
